// GCN_78271484002571
// MI455X (gfx1250) — compile-verified, matrix operand order checked by reading, not run on hardware
//
#include <hip/hip_runtime.h>
#include <hip/hip_bf16.h>
#include <stdint.h>

#define NN 50000
#define NE 1600000
#define CH 256

typedef __attribute__((ext_vector_type(16))) __bf16 v16bf;
typedef __attribute__((ext_vector_type(8)))  float  v8f;

__device__ __forceinline__ float atomic_add_f32(float* p, float v) {
  return unsafeAtomicAdd(p, v);   // lowers to global_atomic_add_f32 on gfx1250
}

// ---------------- degree accumulation: deg[dst] += w ----------------
__global__ void k_deg(const int* __restrict__ ei, const float* __restrict__ ew,
                      float* __restrict__ deg) {
  unsigned e = blockIdx.x * 256u + threadIdx.x;
  if (e < NE) atomic_add_f32(&deg[ei[NE + e]], ew[e]);
}

// ---------------- dinv[i] = rsqrt(deg[i] + 1) (self loop weight 1) --------
__global__ void k_dinv(float* __restrict__ d) {
  unsigned i = blockIdx.x * 256u + threadIdx.x;
  if (i < NN) d[i] = rsqrtf(d[i] + 1.0f);
}

// ---------------- W (256x256 f32) -> bf16 hi/lo WMMA B-fragments ----------
// Fragment order: [kt(8)][nt(16)][lane(32)][16 bf16]; B 32x16 tile layout:
// lane<16: N=lane, K = kt*32 + {0..7, 16..23}; lane>=16: N=lane-16, K = {8..15, 24..31}
__global__ void k_wfrag(const float* __restrict__ W, __bf16* __restrict__ fhi,
                        __bf16* __restrict__ flo) {
  unsigned t = blockIdx.x * 256u + threadIdx.x;     // 8*16*32 = 4096 threads
  if (t >= 8u * 16u * 32u) return;
  int lane = t & 31, tile = t >> 5;
  int nt = tile & 15, kt = tile >> 4;
  int n = nt * 16 + (lane & 15);
  int khalf = lane >> 4;
  size_t base = (size_t)t * 16;
#pragma unroll
  for (int e = 0; e < 16; ++e) {
    int k = kt * 32 + ((e < 8) ? (khalf * 8 + e) : (16 + khalf * 8 + (e - 8)));
    float v = W[(size_t)k * CH + n];
    __bf16 hi = (__bf16)v;
    fhi[base + e] = hi;
    flo[base + e] = (__bf16)(v - (float)hi);
  }
}

// ---------------- GEMM: Cout[NN x 256] = A[NN x 256] * W  (split-bf16 WMMA) -
__global__ __launch_bounds__(256) void k_gemm(const float* __restrict__ A,
                                              const __bf16* __restrict__ Whi,
                                              const __bf16* __restrict__ Wlo,
                                              float* __restrict__ Cout) {
  const int tid  = threadIdx.x;
  const int lane = tid & 31, wave = tid >> 5;
  const int rg = wave >> 2, cg = wave & 3;          // 2 row-groups x 4 col-groups
  const int m0 = blockIdx.x * 32 + rg * 16;
  const int n0 = cg * 64;
  const int mA = m0 + (lane & 15);
  const int khalf = lane >> 4;
  const bool arow_ok = (mA < NN);
  const float* arow = A + (size_t)(arow_ok ? mA : (NN - 1)) * CH;

  v8f acc0 = {0,0,0,0,0,0,0,0};
  v8f acc1 = acc0, acc2 = acc0, acc3 = acc0;

  for (int kt = 0; kt < 8; ++kt) {
    // A fragment: 16 f32 per lane -> bf16 hi/lo (layout per ISA 7.12.2)
    const float4* p = (const float4*)(arow + kt * 32 + khalf * 8);
    float4 f0 = p[0], f1 = p[1];   // K = kt*32 + khalf*8 + [0..7]
    float4 f2 = p[4], f3 = p[5];   // K = kt*32 + 16 + khalf*8 + [0..7]
    float av[16] = {f0.x,f0.y,f0.z,f0.w, f1.x,f1.y,f1.z,f1.w,
                    f2.x,f2.y,f2.z,f2.w, f3.x,f3.y,f3.z,f3.w};
    v16bf ahi, alo;
#pragma unroll
    for (int e = 0; e < 16; ++e) {
      float v = arow_ok ? av[e] : 0.0f;
      __bf16 hi = (__bf16)v;
      ahi[e] = hi;
      alo[e] = (__bf16)(v - (float)hi);
    }
    v8f* accs[4] = {&acc0, &acc1, &acc2, &acc3};
#pragma unroll
    for (int t = 0; t < 4; ++t) {
      int nt = cg * 4 + t;
      size_t fb = ((size_t)(kt * 16 + nt) * 32 + lane) * 16;
      v16bf bhi = *(const v16bf*)(Whi + fb);
      v16bf blo = *(const v16bf*)(Wlo + fb);
      v8f a = *accs[t];
      a = __builtin_amdgcn_wmma_f32_16x16x32_bf16(false, ahi, false, bhi, (short)0, a, false, false);
      a = __builtin_amdgcn_wmma_f32_16x16x32_bf16(false, ahi, false, blo, (short)0, a, false, false);
      a = __builtin_amdgcn_wmma_f32_16x16x32_bf16(false, alo, false, bhi, (short)0, a, false, false);
      *accs[t] = a;
    }
  }

  // Store: C/D layout -> element r: M = m0 + (lane>=16 ? 8 : 0) + r, N = n0 + t*16 + lane%16
  const int mc0 = m0 + khalf * 8;
  const int nc  = lane & 15;
  v8f* accs[4] = {&acc0, &acc1, &acc2, &acc3};
#pragma unroll
  for (int t = 0; t < 4; ++t) {
    int n = n0 + t * 16 + nc;
#pragma unroll
    for (int r = 0; r < 8; ++r) {
      int m = mc0 + r;
      if (m < NN) Cout[(size_t)m * CH + n] = (*accs[t])[r];
    }
  }
}

// ---------------- edge scatter: agg[dst] += H[src] * dinv[s]*w*dinv[d] ------
__global__ void k_scatter(const int* __restrict__ ei, const float* __restrict__ ew,
                          const float* __restrict__ dinv, const float* __restrict__ H,
                          float* __restrict__ agg) {
  unsigned t = blockIdx.x * 256u + threadIdx.x;   // 64 threads per edge
  unsigned e = t >> 6;
  if (e >= NE) return;
  int c = (t & 63) << 2;
  int s = ei[e], d = ei[NE + e];
  float nrm = dinv[s] * ew[e] * dinv[d];
  float4 h = *(const float4*)(H + (size_t)s * CH + c);
  float* o = agg + (size_t)d * CH + c;
  atomic_add_f32(o + 0, h.x * nrm);
  atomic_add_f32(o + 1, h.y * nrm);
  atomic_add_f32(o + 2, h.z * nrm);
  atomic_add_f32(o + 3, h.w * nrm);
}

// --------- epilogue: act = relu(agg + H*dinv^2 (self loop) + bias) ----------
__global__ void k_epilogue(const float* __restrict__ agg, const float* __restrict__ H,
                           const float* __restrict__ dinv, const float* __restrict__ b,
                           float* __restrict__ act) {
  unsigned t = blockIdx.x * 256u + threadIdx.x;   // NN*64 float4 slots
  if (t >= (unsigned)NN * 64u) return;
  int i = t >> 6, c = (t & 63) << 2;
  float di = dinv[i];
  float sl = di * di;
  float4 a  = *(const float4*)(agg + (size_t)i * CH + c);
  float4 h  = *(const float4*)(H   + (size_t)i * CH + c);
  float4 bb = *(const float4*)(b + c);
  float4 r;
  r.x = fmaxf(a.x + h.x * sl + bb.x, 0.0f);
  r.y = fmaxf(a.y + h.y * sl + bb.y, 0.0f);
  r.z = fmaxf(a.z + h.z * sl + bb.z, 0.0f);
  r.w = fmaxf(a.w + h.w * sl + bb.w, 0.0f);
  *(float4*)(act + (size_t)i * CH + c) = r;
}

// ---------------- head: out[i] = dot(H[i], Wf) + bf  (wave32 per node) ------
__global__ void k_head(const float* __restrict__ H, const float* __restrict__ Wf,
                       const float* __restrict__ bfp, float* __restrict__ out) {
  int lane = threadIdx.x & 31, wave = threadIdx.x >> 5;
  int i = blockIdx.x * 8 + wave;
  if (i >= NN) return;
  const float4* hp = (const float4*)(H + (size_t)i * CH) + lane * 2;
  const float4* wp = (const float4*)Wf + lane * 2;
  float4 h0 = hp[0], h1 = hp[1], w0 = wp[0], w1 = wp[1];
  float s = h0.x * w0.x + h0.y * w0.y + h0.z * w0.z + h0.w * w0.w +
            h1.x * w1.x + h1.y * w1.y + h1.z * w1.z + h1.w * w1.w;
#pragma unroll
  for (int off = 16; off > 0; off >>= 1) s += __shfl_xor(s, off, 32);
  if (lane == 0) out[i] = s + bfp[0];
}

extern "C" void kernel_launch(void* const* d_in, const int* in_sizes, int n_in,
                              void* d_out, int out_size, void* d_ws, size_t ws_size,
                              hipStream_t stream) {
  const float* x  = (const float*)d_in[0];
  const int*   ei = (const int*)  d_in[1];
  const float* ew = (const float*)d_in[2];
  const float* W1 = (const float*)d_in[3];
  const float* b1 = (const float*)d_in[4];
  const float* W2 = (const float*)d_in[5];
  const float* b2 = (const float*)d_in[6];
  const float* Wf = (const float*)d_in[7];
  const float* bf = (const float*)d_in[8];
  float* out = (float*)d_out;

  char* ws = (char*)d_ws;
  size_t off = 0;
  auto take = [&](size_t bytes) { char* p = ws + off; off += (bytes + 255) & ~(size_t)255; return p; };
  float*  dinv = (float*)take((size_t)NN * 4);
  __bf16* w1hi = (__bf16*)take(8 * 16 * 32 * 16 * 2);
  __bf16* w1lo = (__bf16*)take(8 * 16 * 32 * 16 * 2);
  __bf16* w2hi = (__bf16*)take(8 * 16 * 32 * 16 * 2);
  __bf16* w2lo = (__bf16*)take(8 * 16 * 32 * 16 * 2);
  float*  Hb   = (float*)take((size_t)NN * CH * 4);
  float*  AGG  = (float*)take((size_t)NN * CH * 4);
  float*  ACT  = (float*)take((size_t)NN * CH * 4);
  (void)ws_size; (void)in_sizes; (void)n_in; (void)out_size;

  const int gemm_blocks = (NN + 31) / 32;                 // 1563
  const unsigned scat_blocks = (unsigned)(((size_t)NE * 64 + 255) / 256);  // 400000
  const int epi_blocks = ((NN * 64) + 255) / 256;         // 12500

  // normalization
  hipMemsetAsync(dinv, 0, (size_t)NN * 4, stream);
  k_deg<<<(NE + 255) / 256, 256, 0, stream>>>(ei, ew, dinv);
  k_dinv<<<(NN + 255) / 256, 256, 0, stream>>>(dinv);

  // weight fragment precompute (bf16 hi/lo split)
  k_wfrag<<<16, 256, 0, stream>>>(W1, w1hi, w1lo);
  k_wfrag<<<16, 256, 0, stream>>>(W2, w2hi, w2lo);

  // layer 1
  k_gemm<<<gemm_blocks, 256, 0, stream>>>(x, w1hi, w1lo, Hb);
  hipMemsetAsync(AGG, 0, (size_t)NN * CH * 4, stream);
  k_scatter<<<scat_blocks, 256, 0, stream>>>(ei, ew, dinv, Hb, AGG);
  k_epilogue<<<epi_blocks, 256, 0, stream>>>(AGG, Hb, dinv, b1, ACT);

  // layer 2
  k_gemm<<<gemm_blocks, 256, 0, stream>>>(ACT, w2hi, w2lo, Hb);
  hipMemsetAsync(AGG, 0, (size_t)NN * CH * 4, stream);
  k_scatter<<<scat_blocks, 256, 0, stream>>>(ei, ew, dinv, Hb, AGG);
  k_epilogue<<<epi_blocks, 256, 0, stream>>>(AGG, Hb, dinv, b2, ACT);

  // regression head
  k_head<<<(NN + 7) / 8, 256, 0, stream>>>(ACT, Wf, bf, out);
}
